// Decoder_84610855731706
// MI455X (gfx1250) — matrix operand order repaired and checked by reading, not compiled, not run on hardware
//
#include <hip/hip_runtime.h>
#include <hip/hip_bf16.h>
#include <math.h>

typedef __attribute__((ext_vector_type(16))) __bf16 v16bf;
typedef __attribute__((ext_vector_type(8)))  float  v8f;

#define BATCH 256
#define TSTEPS 40
#define VOCAB 5000
#define DD 512
#define ZZ 128
#define NPAD 5056   // vocab padded to multiple of 64 (block N tile)

struct Frag { union { v16bf v; uint4 q[2]; }; };

// ---------------------------------------------------------------------------
// Plain WMMA bf16 GEMM:  C[M=256, N] = A[M,K] * B[N,K]^T + bias
// Block = 8 waves stacked along M (block tile 256M x 64N), grid = (N/64, 1):
// every B fragment is unique per block -> weight bytes hit L2 once per WGP,
// intra-block reuse is WGP$-resident. Per wave: 2 A-frags x 4 B-frags = 8
// WMMAs per 32-wide K step, register double-buffered (K % 64 == 0).
// MODE: 0 = store f32, 1 = relu -> bf16, 2 = tanh -> f32
// ---------------------------------------------------------------------------
template <int MODE>
__global__ void __launch_bounds__(256)
wmma_gemm(const __bf16* __restrict__ A, int lda,
          const __bf16* __restrict__ B, int ldb,
          const float* __restrict__ bias, int nbias,
          void* __restrict__ Cout, int ldc, int K)
{
    const int wave = threadIdx.x >> 5;
    const int lane = threadIdx.x & 31;
    const int half = lane >> 4;
    const int l16  = lane & 15;
    const int mbase = wave * 32;
    const int nbase = blockIdx.x * 64;

    const __bf16* Arow0 = A + (size_t)(mbase + l16) * lda;
    const __bf16* Arow1 = Arow0 + (size_t)16 * lda;
    const __bf16* Bbase = B + (size_t)(nbase + l16) * ldb + half * 8;

    v8f acc[2][4];
#pragma unroll
    for (int j = 0; j < 4; ++j) {
        const int n = nbase + j * 16 + l16;
        const float bv = (bias && n < nbias) ? bias[n] : 0.f;
        acc[0][j] = (v8f){bv, bv, bv, bv, bv, bv, bv, bv};
        acc[1][j] = acc[0][j];
    }

    Frag a0[2], a1[2], b0[4], b1[4];
    auto loadA = [&](Frag (&a)[2], int k) {
        const char* p0 = (const char*)(Arow0 + k + half * 8);
        a[0].q[0] = *(const uint4*)(p0);
        a[0].q[1] = *(const uint4*)(p0 + 32);
        const char* p1 = (const char*)(Arow1 + k + half * 8);
        a[1].q[0] = *(const uint4*)(p1);
        a[1].q[1] = *(const uint4*)(p1 + 32);
    };
    auto loadB = [&](Frag (&bf)[4], int k) {
#pragma unroll
        for (int j = 0; j < 4; ++j) {
            const uint4* p = (const uint4*)(Bbase + (size_t)(j * 16) * ldb + k);
            bf[j].q[0] = p[0];
            bf[j].q[1] = p[2];
        }
    };
    auto mm = [&](Frag (&a)[2], Frag (&bf)[4]) {
#pragma unroll
        for (int i = 0; i < 2; ++i)
#pragma unroll
            for (int j = 0; j < 4; ++j)
                acc[i][j] = __builtin_amdgcn_wmma_f32_16x16x32_bf16(
                    false, a[i].v, false, bf[j].v, (short)0, acc[i][j], false, false);
    };

    loadA(a0, 0); loadB(b0, 0);
    int k = 0;
    for (; k + 64 < K; k += 64) {
        loadA(a1, k + 32); loadB(b1, k + 32);
        mm(a0, b0);
        loadA(a0, k + 64); loadB(b0, k + 64);
        mm(a1, b1);
    }
    loadA(a1, k + 32); loadB(b1, k + 32);
    mm(a0, b0);
    mm(a1, b1);

#pragma unroll
    for (int i = 0; i < 2; ++i) {
#pragma unroll
        for (int j = 0; j < 4; ++j) {
            const int n = nbase + j * 16 + l16;
#pragma unroll
            for (int r = 0; r < 8; ++r) {
                const float v = acc[i][j][r];
                const size_t idx = (size_t)(mbase + i * 16 + r + half * 8) * ldc + n;
                if (MODE == 0)      ((float*)Cout)[idx]  = v;
                else if (MODE == 1) ((__bf16*)Cout)[idx] = (__bf16)fmaxf(v, 0.f);
                else                ((float*)Cout)[idx]  = tanhf(v);
            }
        }
    }
}

// ---------------------------------------------------------------------------
// Fused LSTM gate GEMM + pointwise cell update.
// Each block computes a 16-unit slice across ALL 4 gates: B fragment j covers
// weight rows j*512 + [u0, u0+16). After the K loop each lane holds i/f/o/g
// for its (m, u) elements -> apply sigmoid/tanh, update c, emit bf16 h.
// LAYER 0: h -> h0f (f32), d1 = next-step A0 h-slot, d2 = current A1 h0-slot.
// LAYER 1: d1 = next-step A1 h1-slot, d2 = AL (stores bf16(h + h0f)).
// Grid = (512/16 = 32, 1); block = 8 waves stacked along M.
// ---------------------------------------------------------------------------
template <int LAYER>
__global__ void __launch_bounds__(256)
wmma_lstm(const __bf16* __restrict__ A, int lda,
          const __bf16* __restrict__ B, int ldb,
          const float* __restrict__ bias,
          float* __restrict__ c, float* __restrict__ h0f,
          __bf16* __restrict__ d1, int ld1,
          __bf16* __restrict__ d2, int ld2, int K)
{
    const int wave = threadIdx.x >> 5;
    const int lane = threadIdx.x & 31;
    const int half = lane >> 4;
    const int l16  = lane & 15;
    const int mbase = wave * 32;
    const int u0    = blockIdx.x * 16;

    const __bf16* Arow0 = A + (size_t)(mbase + l16) * lda;
    const __bf16* Arow1 = Arow0 + (size_t)16 * lda;
    const __bf16* Bbase = B + (size_t)(u0 + l16) * ldb + half * 8;  // gate 0 row

    v8f acc[2][4];
#pragma unroll
    for (int j = 0; j < 4; ++j) {
        const float bv = bias[j * 512 + u0 + l16];
        acc[0][j] = (v8f){bv, bv, bv, bv, bv, bv, bv, bv};
        acc[1][j] = acc[0][j];
    }

    Frag a0[2], a1[2], b0[4], b1[4];
    auto loadA = [&](Frag (&a)[2], int k) {
        const char* p0 = (const char*)(Arow0 + k + half * 8);
        a[0].q[0] = *(const uint4*)(p0);
        a[0].q[1] = *(const uint4*)(p0 + 32);
        const char* p1 = (const char*)(Arow1 + k + half * 8);
        a[1].q[0] = *(const uint4*)(p1);
        a[1].q[1] = *(const uint4*)(p1 + 32);
    };
    auto loadB = [&](Frag (&bf)[4], int k) {
#pragma unroll
        for (int j = 0; j < 4; ++j) {   // j = gate index, stride 512 rows
            const uint4* p = (const uint4*)(Bbase + (size_t)(j * 512) * ldb + k);
            bf[j].q[0] = p[0];
            bf[j].q[1] = p[2];
        }
    };
    auto mm = [&](Frag (&a)[2], Frag (&bf)[4]) {
#pragma unroll
        for (int i = 0; i < 2; ++i)
#pragma unroll
            for (int j = 0; j < 4; ++j)
                acc[i][j] = __builtin_amdgcn_wmma_f32_16x16x32_bf16(
                    false, a[i].v, false, bf[j].v, (short)0, acc[i][j], false, false);
    };

    loadA(a0, 0); loadB(b0, 0);
    int k = 0;
    for (; k + 64 < K; k += 64) {
        loadA(a1, k + 32); loadB(b1, k + 32);
        mm(a0, b0);
        loadA(a0, k + 64); loadB(b0, k + 64);
        mm(a1, b1);
    }
    loadA(a1, k + 32); loadB(b1, k + 32);
    mm(a0, b0);
    mm(a1, b1);

    // fused LSTM pointwise epilogue
    const int u = u0 + l16;
#pragma unroll
    for (int i = 0; i < 2; ++i) {
#pragma unroll
        for (int r = 0; r < 8; ++r) {
            const int m = mbase + i * 16 + r + half * 8;
            const size_t cu = (size_t)m * 512 + u;
            float ig = 1.f / (1.f + __expf(-acc[i][0][r]));
            float fg = 1.f / (1.f + __expf(-acc[i][1][r]));
            float og = 1.f / (1.f + __expf(-acc[i][2][r]));
            float cn = tanhf(acc[i][3][r]);
            float cv = fg * c[cu] + ig * cn;
            c[cu] = cv;
            float h = og * tanhf(cv);
            d1[(size_t)m * ld1 + u] = (__bf16)h;
            if (LAYER == 0) {
                h0f[cu] = h;
                d2[(size_t)m * ld2 + u] = (__bf16)h;
            } else {
                d2[(size_t)m * ld2 + u] = (__bf16)(h + h0f[cu]);
            }
        }
    }
}

// ---------------------------------------------------------------------------
// Elementwise / helper kernels
// ---------------------------------------------------------------------------
__global__ void cvt_f32_bf16(const float* __restrict__ s, __bf16* __restrict__ d, int n) {
    int i = blockIdx.x * blockDim.x + threadIdx.x;
    if (i < n) d[i] = (__bf16)s[i];
}

__global__ void cvt_pad_rows(const float* __restrict__ s, __bf16* __restrict__ d,
                             int rows, int cols, int rowsPad) {
    int i = blockIdx.x * blockDim.x + threadIdx.x;
    if (i >= rowsPad * cols) return;
    int r = i / cols;
    d[i] = (r < rows) ? (__bf16)s[i] : (__bf16)0.f;
}

__global__ void zero_f32(float* p, int n) {
    int i = blockIdx.x * blockDim.x + threadIdx.x;
    if (i < n) p[i] = 0.f;
}

// bf16(z) into both A0 buffers [:,1024:1152) and AL[:,512:640)
__global__ void z_scatter(const float* __restrict__ z, __bf16* __restrict__ A0a,
                          __bf16* __restrict__ A0b, __bf16* __restrict__ AL) {
    int i = blockIdx.x * blockDim.x + threadIdx.x;   // 256*128
    int b = i >> 7, cc = i & 127;
    __bf16 v = (__bf16)z[i];
    A0a[b * 1152 + 1024 + cc] = v;
    A0b[b * 1152 + 1024 + cc] = v;
    AL[b * 640 + 512 + cc]    = v;
}

// split init hh=(h|c): h -> bf16 slot of buffer 0, c -> f32 state
__global__ void init_scatter(const float* __restrict__ hh, float* __restrict__ cbuf,
                             __bf16* __restrict__ hdst, int ldh) {
    int i = blockIdx.x * blockDim.x + threadIdx.x;   // 256*512
    int b = i >> 9, d = i & 511;
    hdst[b * ldh + d] = (__bf16)hh[b * 1024 + d];
    cbuf[i] = hh[b * 1024 + 512 + d];
}

// gather bf16 embedding row for token t into A0[:,512:1024) (t = 0 only)
__global__ void embed_gather(const __bf16* __restrict__ embb, const int* __restrict__ x,
                             int t, __bf16* __restrict__ A0) {
    int i = blockIdx.x * blockDim.x + threadIdx.x;   // 256*512
    int b = i >> 9, d = i & 511;
    int tok = x[b * TSTEPS + t];
    A0[b * 1152 + 512 + d] = embb[(size_t)tok * DD + d];
}

// per-row log-softmax + NLL gather accumulated into out[b]; optionally also
// gathers this step's embedding row into A0's e-slot for the following multi.
__global__ void softmax_nll(const float* __restrict__ logits, int ldl,
                            const int* __restrict__ x, int t, float* __restrict__ out,
                            const __bf16* __restrict__ embb, __bf16* __restrict__ A0,
                            int doGather) {
    __shared__ float red[256];
    int b = blockIdx.x;
    const float* row = logits + (size_t)b * ldl;
    float mx = -INFINITY;
    for (int j = threadIdx.x; j < VOCAB; j += 256) mx = fmaxf(mx, row[j]);
    red[threadIdx.x] = mx; __syncthreads();
    for (int s = 128; s > 0; s >>= 1) {
        if (threadIdx.x < s) red[threadIdx.x] = fmaxf(red[threadIdx.x], red[threadIdx.x + s]);
        __syncthreads();
    }
    mx = red[0]; __syncthreads();
    float sum = 0.f;
    for (int j = threadIdx.x; j < VOCAB; j += 256) sum += __expf(row[j] - mx);
    red[threadIdx.x] = sum; __syncthreads();
    for (int s = 128; s > 0; s >>= 1) {
        if (threadIdx.x < s) red[threadIdx.x] += red[threadIdx.x + s];
        __syncthreads();
    }
    int tok = x[b * TSTEPS + t];
    if (threadIdx.x == 0) out[b] += row[tok] - mx - __logf(red[0]);
    if (doGather) {
        for (int j = threadIdx.x; j < DD; j += 256)
            A0[b * 1152 + 512 + j] = embb[(size_t)tok * DD + j];
    }
}

// ---------------------------------------------------------------------------
extern "C" void kernel_launch(void* const* d_in, const int* in_sizes, int n_in,
                              void* d_out, int out_size, void* d_ws, size_t ws_size,
                              hipStream_t stream) {
    const float* z    = (const float*)d_in[0];
    const int*   x    = (const int*)d_in[1];
    const float* emb  = (const float*)d_in[2];
    const float* Wg0  = (const float*)d_in[3];
    const float* bg0  = (const float*)d_in[4];
    const float* Wg1  = (const float*)d_in[5];
    const float* bg1  = (const float*)d_in[6];
    const float* Wout = (const float*)d_in[7];
    const float* bout = (const float*)d_in[8];
    const float* tw1  = (const float*)d_in[9];
    const float* tb1  = (const float*)d_in[10];
    const float* tw2  = (const float*)d_in[11];
    const float* tb2  = (const float*)d_in[12];
    float* out = (float*)d_out;

    char* ws = (char*)d_ws;
    auto carve = [&](size_t bytes) -> char* {
        char* p = ws; ws += (bytes + 255) & ~(size_t)255; return p;
    };
    __bf16* Wg0b  = (__bf16*)carve(2048ULL * 1152 * 2);
    __bf16* Wg1b  = (__bf16*)carve(2048ULL * 1024 * 2);
    __bf16* Woutb = (__bf16*)carve((size_t)NPAD * 640 * 2);
    __bf16* embb  = (__bf16*)carve(5000ULL * 512 * 2);
    __bf16* tw1b  = (__bf16*)carve(2ULL * 2048 * 128 * 2);
    __bf16* tw2b  = (__bf16*)carve(2ULL * 1024 * 2048 * 2);
    __bf16* A0buf[2], *A1buf[2];
    A0buf[0] = (__bf16*)carve(256ULL * 1152 * 2);   // [h0 | e_t | z] ping
    A0buf[1] = (__bf16*)carve(256ULL * 1152 * 2);   //                pong
    A1buf[0] = (__bf16*)carve(256ULL * 1024 * 2);   // [h1 | h0 ]     ping
    A1buf[1] = (__bf16*)carve(256ULL * 1024 * 2);   //                pong
    __bf16* AL    = (__bf16*)carve(256ULL * 640 * 2);    // [h0+h1 | z]
    __bf16* a1t   = (__bf16*)carve(256ULL * 2048 * 2);   // init relu hidden
    float*  hh    = (float*)carve(256ULL * 1024 * 4);    // init tanh out
    float*  logit = (float*)carve((size_t)256 * NPAD * 4);
    float*  c0    = (float*)carve(256ULL * 512 * 4);
    float*  c1    = (float*)carve(256ULL * 512 * 4);
    float*  h0f   = (float*)carve(256ULL * 512 * 4);

    auto cvt = [&](const float* s, __bf16* d, size_t n) {
        cvt_f32_bf16<<<dim3((unsigned)((n + 255) / 256)), dim3(256), 0, stream>>>(s, d, (int)n);
    };
    cvt(Wg0, Wg0b, 2048ULL * 1152);
    cvt(Wg1, Wg1b, 2048ULL * 1024);
    cvt(emb, embb, 5000ULL * 512);
    cvt(tw1, tw1b, 2ULL * 2048 * 128);
    cvt(tw2, tw2b, 2ULL * 1024 * 2048);
    {
        size_t n = (size_t)NPAD * 640;
        cvt_pad_rows<<<dim3((unsigned)((n + 255) / 256)), dim3(256), 0, stream>>>(
            Wout, Woutb, VOCAB, 640, NPAD);
    }
    z_scatter<<<dim3(128), dim3(256), 0, stream>>>(z, A0buf[0], A0buf[1], AL);
    zero_f32<<<dim3(1), dim3(256), 0, stream>>>(out, 256);

    // ----- initial states: h,c = split(tanh(relu(z @ tw1^T + tb1) @ tw2^T + tb2))
    for (int l = 0; l < 2; ++l) {
        wmma_gemm<1><<<dim3(32), dim3(256), 0, stream>>>(
            A0buf[0] + 1024, 1152, tw1b + (size_t)l * 2048 * 128, 128,
            tb1 + l * 2048, 2048, a1t, 2048, 128);
        wmma_gemm<2><<<dim3(16), dim3(256), 0, stream>>>(
            a1t, 2048, tw2b + (size_t)l * 1024 * 2048, 2048,
            tb2 + l * 1024, 1024, hh, 1024, 2048);
        init_scatter<<<dim3(512), dim3(256), 0, stream>>>(
            hh, (l == 0 ? c0 : c1), (l == 0 ? A0buf[0] : A1buf[0]),
            (l == 0 ? 1152 : 1024));
    }

    // one LSTM step for token parity t: reads ping, writes h into pong
    auto multi = [&](int t) {
        __bf16* A0r = A0buf[t & 1];
        __bf16* A0w = A0buf[(t + 1) & 1];
        __bf16* A1r = A1buf[t & 1];
        __bf16* A1w = A1buf[(t + 1) & 1];
        // layer 0: gates from [h0|e_t|z]; h0 -> next A0 h-slot, current A1 h0-slot
        wmma_lstm<0><<<dim3(32), dim3(256), 0, stream>>>(
            A0r, 1152, Wg0b, 1152, bg0, c0, h0f,
            A0w, 1152, A1r + 512, 1024, 1152);
        // layer 1: gates from [h1|h0]; h1 -> next A1 h1-slot; AL = bf16(h0+h1)
        wmma_lstm<1><<<dim3(32), dim3(256), 0, stream>>>(
            A1r, 1024, Wg1b, 1024, bg1, c1, h0f,
            A1w, 1024, AL, 640, 1024);
    };

    // token 0
    embed_gather<<<dim3(512), dim3(256), 0, stream>>>(embb, x, 0, A0buf[0]);
    multi(0);

    for (int t = 1; t < TSTEPS; ++t) {
        // logits from previous h_t, score token t; fused gather of e_t
        wmma_gemm<0><<<dim3(NPAD / 64), dim3(256), 0, stream>>>(
            AL, 640, Woutb, 640, bout, VOCAB, logit, NPAD, 640);
        softmax_nll<<<dim3(256), dim3(256), 0, stream>>>(
            logit, NPAD, x, t, out, embb, A0buf[t & 1], (t < TSTEPS - 1) ? 1 : 0);
        if (t < TSTEPS - 1) multi(t);    // final multi's output is never used
    }
}
